// ModernBertEncoderLayer_43147241456092
// MI455X (gfx1250) — compile-verified, matrix operand order checked by reading, not run on hardware
//
#include <hip/hip_runtime.h>

// ---------------------------------------------------------------------------
// ModernBERT encoder layer for gfx1250 (MI455X), bf16 WMMA pipeline.
// B=2, S=2048, HIDDEN=768, HEADS=12, HEAD_DIM=64, INTER=1152, WINDOW=±64.
// GEMMs: pre-transposed bf16 weights (N x K, K contiguous), double-buffered
// LDS fed by async global->LDS DMA (ASYNCcnt) when available, BK=64,
// 8 WMMAs / wave / iteration, 1 barrier / iteration.
// ---------------------------------------------------------------------------

typedef __attribute__((ext_vector_type(16))) __bf16 v16bf;
typedef __attribute__((ext_vector_type(8)))  float  v8f;
typedef int v4i32 __attribute__((vector_size(16)));

#if defined(__has_builtin)
#if __has_builtin(__builtin_amdgcn_global_load_async_to_lds_b128)
#define HAVE_ASYNC_LDS 1
#endif
#endif
#ifndef HAVE_ASYNC_LDS
#define HAVE_ASYNC_LDS 0
#endif

static __device__ __forceinline__ __bf16 tobf(float x) { return (__bf16)x; }

// 16-byte global -> LDS copy; async DMA path on gfx1250 toolchains that
// expose the builtin, VGPR round-trip otherwise.
static __device__ __forceinline__ void copy16_g2l(const __bf16* g, __bf16* l) {
#if HAVE_ASYNC_LDS
  typedef __attribute__((address_space(1))) v4i32* gp_t;
  typedef __attribute__((address_space(3))) v4i32* lp_t;
  __builtin_amdgcn_global_load_async_to_lds_b128((gp_t)g, (lp_t)l, 0, 0);
#else
  *reinterpret_cast<uint4*>(l) = *reinterpret_cast<const uint4*>(g);
#endif
}

static __device__ __forceinline__ void wait_g2l() {
#if HAVE_ASYNC_LDS
#if __has_builtin(__builtin_amdgcn_s_wait_asynccnt)
  __builtin_amdgcn_s_wait_asynccnt(0);
#else
  asm volatile("s_wait_asynccnt 0x0" ::: "memory");
#endif
#endif
}

static __device__ __forceinline__ v8f wmma_bf16(v16bf a, v16bf b, v8f c) {
  return __builtin_amdgcn_wmma_f32_16x16x32_bf16(
      /*neg_a=*/false, a, /*neg_b=*/false, b,
      /*c_mod=*/(short)0, c, /*reuse_a=*/false, /*reuse_b=*/false);
}

// 16x32 bf16 A-style fragment (ISA 7.12.2): lane r = row, consecutive-K pairs
// per VGPR; lanes 16-31 take K+8 / K+24. Also used for B when the operand is
// stored transposed (rows = N columns, contiguous along K).
static __device__ __forceinline__ void load_frag(v16bf& f, const __bf16* base,
                                                 int ld, int lane) {
  const int r  = lane & 15;
  const int hi = lane >> 4;
  const __bf16* row = base + (size_t)r * ld;
#pragma unroll
  for (int v = 0; v < 4; ++v) {
    int kb = 2 * v + hi * 8;
    f[2 * v]     = row[kb];
    f[2 * v + 1] = row[kb + 1];
  }
#pragma unroll
  for (int v = 0; v < 4; ++v) {
    int kb = 16 + 2 * v + hi * 8;
    f[8 + 2 * v]     = row[kb];
    f[8 + 2 * v + 1] = row[kb + 1];
  }
}

// Per-lane-row variant for QK^T B-fragments gathered directly from global K
// rows (lane = key column, pairs along head-dim). Zero-fills invalid keys.
static __device__ __forceinline__ void load_rowfrag(v16bf& f, const __bf16* row,
                                                    bool valid, int hi) {
#pragma unroll
  for (int v = 0; v < 8; ++v) {
    int kb = ((v < 4) ? (2 * v) : (16 + 2 * (v - 4))) + hi * 8;
    f[2 * v]     = valid ? row[kb]     : tobf(0.f);
    f[2 * v + 1] = valid ? row[kb + 1] : tobf(0.f);
  }
}

// ---------------------------------------------------------------------------
// f32[K,N] -> bf16[N,K] tiled transpose (weights; one-time cost, 64x reuse)
// ---------------------------------------------------------------------------
__global__ __launch_bounds__(256) void cvt_transpose_kernel(
    const float* __restrict__ src, __bf16* __restrict__ dst, int K, int N) {
  __shared__ __bf16 t[32][33];
  const int kb = blockIdx.y * 32;
  const int nb = blockIdx.x * 32;
  const int tx = threadIdx.x;  // 32
  const int ty = threadIdx.y;  // 8
#pragma unroll
  for (int r = ty; r < 32; r += 8)
    t[r][tx] = tobf(src[(size_t)(kb + r) * N + nb + tx]);
  __syncthreads();
#pragma unroll
  for (int r = ty; r < 32; r += 8)
    dst[(size_t)(nb + r) * K + kb + tx] = t[tx][r];
}

// ---------------------------------------------------------------------------
// RMSNorm (per 768-wide row) -> bf16
// ---------------------------------------------------------------------------
__global__ __launch_bounds__(256) void rmsnorm_kernel(
    const float* __restrict__ x, const float* __restrict__ g,
    __bf16* __restrict__ o) {
  const int row = blockIdx.x;
  const int tid = threadIdx.x;
  const float* xr = x + (size_t)row * 768;
  float s = 0.f;
#pragma unroll
  for (int j = tid; j < 768; j += 256) s += xr[j] * xr[j];
  __shared__ float red[256];
  red[tid] = s;
  __syncthreads();
  for (int st = 128; st > 0; st >>= 1) {
    if (tid < st) red[tid] += red[tid + st];
    __syncthreads();
  }
  const float inv = rsqrtf(red[0] * (1.f / 768.f) + 1e-5f);
#pragma unroll
  for (int j = tid; j < 768; j += 256)
    o[(size_t)row * 768 + j] = tobf(xr[j] * inv * g[j]);
}

// ---------------------------------------------------------------------------
// Tiled bf16 GEMM: C[M,N] = A[M,K] * Bt[N,K]^T (+resid), f32 accumulate.
// 64x64 block tile, BK=64, 4 waves, 2x2 WMMA tiles per wave, double-buffered,
// async global->LDS staging.
// ---------------------------------------------------------------------------
static __device__ __forceinline__ void gemm_stage(
    const __bf16* __restrict__ A, const __bf16* __restrict__ Bt, int m0, int n0,
    int K, int kt, int tid, __bf16* sA, __bf16* sB) {
  const int row = tid >> 1;
  const int kk  = (tid & 1) * 32;
  const __bf16* ga = A + (size_t)(m0 + row) * K + kt + kk;
  const __bf16* gb = Bt + (size_t)(n0 + row) * K + kt + kk;
  __bf16* la = sA + tid * 32;
  __bf16* lb = sB + tid * 32;
#pragma unroll
  for (int j = 0; j < 4; ++j) copy16_g2l(ga + j * 8, la + j * 8);
#pragma unroll
  for (int j = 0; j < 4; ++j) copy16_g2l(gb + j * 8, lb + j * 8);
}

static __device__ __forceinline__ void gemm_compute(
    const __bf16* sA, const __bf16* sB, int wm, int wn, int lane,
    v8f acc[2][2]) {
#pragma unroll
  for (int c = 0; c < 2; ++c) {
    v16bf a0, a1, b0, b1;
    load_frag(a0, sA + (wm + 0)  * 64 + c * 32, 64, lane);
    load_frag(a1, sA + (wm + 16) * 64 + c * 32, 64, lane);
    load_frag(b0, sB + (wn + 0)  * 64 + c * 32, 64, lane);
    load_frag(b1, sB + (wn + 16) * 64 + c * 32, 64, lane);
    acc[0][0] = wmma_bf16(a0, b0, acc[0][0]);
    acc[0][1] = wmma_bf16(a0, b1, acc[0][1]);
    acc[1][0] = wmma_bf16(a1, b0, acc[1][0]);
    acc[1][1] = wmma_bf16(a1, b1, acc[1][1]);
  }
}

__global__ __launch_bounds__(128) void gemm_bf16_kernel(
    const __bf16* __restrict__ A, const __bf16* __restrict__ Bt,
    const float* __restrict__ resid, float* __restrict__ C,
    int M, int N, int K) {
  __shared__ __bf16 sA[2][64 * 64];
  __shared__ __bf16 sB[2][64 * 64];

  const int tid  = threadIdx.x;
  const int lane = tid & 31;
  const int wave = tid >> 5;
  const int m0 = blockIdx.y * 64;
  const int n0 = blockIdx.x * 64;
  const int wm = (wave >> 1) * 32;
  const int wn = (wave & 1) * 32;

  v8f acc[2][2] = {};

  gemm_stage(A, Bt, m0, n0, K, 0, tid, sA[0], sB[0]);
  wait_g2l();
  __syncthreads();

  int p = 0;
  for (int kt = 64; kt < K; kt += 64) {
    // async-issue next tile into the other buffer; DMA overlaps the WMMAs
    gemm_stage(A, Bt, m0, n0, K, kt, tid, sA[p ^ 1], sB[p ^ 1]);
    gemm_compute(sA[p], sB[p], wm, wn, lane, acc);
    wait_g2l();
    __syncthreads();
    p ^= 1;
  }
  gemm_compute(sA[p], sB[p], wm, wn, lane, acc);

  // --- epilogue: C layout lanes 0-15 -> M=r, lanes 16-31 -> M=8+r ---
  const int hi  = lane >> 4;
  const int nlo = lane & 15;
#pragma unroll
  for (int i = 0; i < 2; ++i)
#pragma unroll
    for (int j = 0; j < 2; ++j)
#pragma unroll
      for (int r = 0; r < 8; ++r) {
        const int m = m0 + wm + i * 16 + r + 8 * hi;
        const int n = n0 + wn + j * 16 + nlo;
        const size_t idx = (size_t)m * N + n;
        float v = acc[i][j][r];
        if (resid) v += resid[idx];
        C[idx] = v;
      }
}

// ---------------------------------------------------------------------------
// RoPE + scale + bf16 repack: qkv f32 [B*S, 3*768] -> q/k/v bf16 [B,H,S,D]
// ---------------------------------------------------------------------------
__global__ __launch_bounds__(256) void rope_kernel(
    const float* __restrict__ qkv, __bf16* __restrict__ qo,
    __bf16* __restrict__ ko, __bf16* __restrict__ vo, int S) {
  const int idx = blockIdx.x * blockDim.x + threadIdx.x;  // (b*S+s)*12 + h
  const int h = idx % 12;
  const int t = idx / 12;  // b*S + s
  const int s = t % S;
  const int b = t / S;
  const float* base = qkv + (size_t)t * 2304 + h * 64;
  const size_t ro = ((size_t)(b * 12 + h) * S + s) * 64;
  __bf16* qr = qo + ro;
  __bf16* kr = ko + ro;
  __bf16* vr = vo + ro;
#pragma unroll
  for (int d = 0; d < 32; ++d) {
    // 10000^(-d/32) = exp2(-d * log2(10000)/32)
    const float invf = exp2f(-(float)d * 0.4152410118609203f);
    float si, co;
    __sincosf((float)s * invf, &si, &co);
    const float q1 = base[d], q2 = base[d + 32];
    qr[d]      = tobf((q1 * co - q2 * si) * 0.125f);  // fold 1/sqrt(64)
    qr[d + 32] = tobf((q2 * co + q1 * si) * 0.125f);
    const float k1 = base[768 + d], k2 = base[768 + d + 32];
    kr[d]      = tobf(k1 * co - k2 * si);
    kr[d + 32] = tobf(k2 * co + k1 * si);
  }
#pragma unroll
  for (int d = 0; d < 64; ++d) vr[d] = tobf(base[1536 + d]);
}

// ---------------------------------------------------------------------------
// Windowed attention: per (64-query block, head, batch). Keys in ±64 window
// => 192-wide key range per 64-query block. QK^T and P*V both via WMMA.
// ---------------------------------------------------------------------------
#define LDP 200  // padded LDS row stride (bf16 elems), even => dword pairs ok

__global__ __launch_bounds__(128) void attn_kernel(
    const __bf16* __restrict__ q, const __bf16* __restrict__ k,
    const __bf16* __restrict__ v, const int* __restrict__ pad,
    __bf16* __restrict__ out, int S) {
  __shared__ __bf16 vt[64 * LDP];        // V^T: [d][key] over 192 keys
  __shared__ __bf16 pbuf[4 * 16 * LDP];  // per-wave P (16 x 192)

  const int tid = threadIdx.x;
  const int qb = blockIdx.x, h = blockIdx.y, b = blockIdx.z;
  const int kstart = qb * 64 - 64;
  const size_t ho = (size_t)(b * 12 + h) * S * 64;
  const __bf16* qh = q + ho;
  const __bf16* kh = k + ho;
  const __bf16* vh = v + ho;

  // stage V^T in LDS (zero outside [0,S)); vectorized b128 global reads
  union { uint4 u; __bf16 hv[8]; } vload;
  for (int i = tid; i < (64 * 192) / 8; i += 128) {
    const int e  = i * 8;
    const int t  = e >> 6;        // key (each key row is 64 elems = 8 uint4)
    const int d0 = e & 63;        // head-dim start (multiple of 8)
    const int g  = kstart + t;
    if (g >= 0 && g < S) {
      vload.u = *reinterpret_cast<const uint4*>(vh + (size_t)g * 64 + d0);
    } else {
      vload.u = make_uint4(0, 0, 0, 0);
    }
#pragma unroll
    for (int j = 0; j < 8; ++j) vt[(d0 + j) * LDP + t] = vload.hv[j];
  }
  __syncthreads();

  const int lane = tid & 31;
  const int wave = tid >> 5;
  const int q0 = qb * 64 + wave * 16;
  const int nlo = lane & 15;
  const int hi  = lane >> 4;

  // Q fragments straight from global (rows = queries, contiguous head-dim)
  v16bf a0, a1;
  load_frag(a0, qh + (size_t)q0 * 64,      64, lane);
  load_frag(a1, qh + (size_t)q0 * 64 + 32, 64, lane);

  // scores: 12 key-tiles of 16, K-dim 64 split into two 32-chunks
  v8f sc[12] = {};
#pragma unroll
  for (int kt = 0; kt < 12; ++kt) {
    const int g = kstart + kt * 16 + nlo;  // this lane's key column
    const bool kv = (g >= 0 && g < S);
    const __bf16* krow = kh + (size_t)(kv ? g : 0) * 64;
    v16bf b0, b1;
    load_rowfrag(b0, krow,      kv, hi);
    load_rowfrag(b1, krow + 32, kv, hi);
    sc[kt] = wmma_bf16(a0, b0, sc[kt]);
    sc[kt] = wmma_bf16(a1, b1, sc[kt]);
  }

  // mask (window + padding + bounds) and row-max
  const int* pmb = pad + b * S;
  float rmax[8];
#pragma unroll
  for (int i = 0; i < 8; ++i) rmax[i] = -1e30f;
#pragma unroll
  for (int kt = 0; kt < 12; ++kt) {
    const int g = kstart + kt * 16 + nlo;
    bool kv = (g >= 0 && g < S);
    if (kv) kv = (pmb[g] != 0);
#pragma unroll
    for (int i = 0; i < 8; ++i) {
      const int qg = q0 + i + 8 * hi;
      const int dd = qg - g;
      const bool ok = kv && dd <= 64 && dd >= -64;
      const float sv = ok ? sc[kt][i] : -1e30f;
      sc[kt][i] = sv;
      rmax[i] = fmaxf(rmax[i], sv);
    }
  }
#pragma unroll
  for (int i = 0; i < 8; ++i) {
#pragma unroll
    for (int off = 1; off < 16; off <<= 1)
      rmax[i] = fmaxf(rmax[i], __shfl_xor(rmax[i], off, 32));
  }
  float rsum[8] = {0.f, 0.f, 0.f, 0.f, 0.f, 0.f, 0.f, 0.f};
#pragma unroll
  for (int kt = 0; kt < 12; ++kt)
#pragma unroll
    for (int i = 0; i < 8; ++i) {
      const float e = __expf(sc[kt][i] - rmax[i]);
      sc[kt][i] = e;
      rsum[i] += e;
    }
  float rinv[8];
#pragma unroll
  for (int i = 0; i < 8; ++i) {
#pragma unroll
    for (int off = 1; off < 16; off <<= 1)
      rsum[i] += __shfl_xor(rsum[i], off, 32);
    rinv[i] = 1.f / rsum[i];
  }

  // stash P (unnormalized) to LDS in A-fragment-friendly layout
  __bf16* pw = pbuf + wave * 16 * LDP;
#pragma unroll
  for (int kt = 0; kt < 12; ++kt)
#pragma unroll
    for (int i = 0; i < 8; ++i) {
      const int m = i + 8 * hi;
      pw[m * LDP + kt * 16 + nlo] = tobf(sc[kt][i]);
    }
  __syncthreads();

  // out[16x64] = P[16x192] @ V[192x64]
  v8f o[4] = {};
#pragma unroll
  for (int kc = 0; kc < 6; ++kc) {
    v16bf pa;
    load_frag(pa, pw + kc * 32, LDP, lane);
#pragma unroll
    for (int dt = 0; dt < 4; ++dt) {
      v16bf vb;
      load_frag(vb, vt + (dt * 16) * LDP + kc * 32, LDP, lane);
      o[dt] = wmma_bf16(pa, vb, o[dt]);
    }
  }

  // write [B*S, 768] bf16, col = h*64 + d, normalize rows
#pragma unroll
  for (int dt = 0; dt < 4; ++dt)
#pragma unroll
    for (int i = 0; i < 8; ++i) {
      const int m = i + 8 * hi;
      const size_t row = (size_t)b * S + q0 + m;
      out[row * 768 + h * 64 + dt * 16 + nlo] = tobf(o[dt][i] * rinv[i]);
    }
}

// ---------------------------------------------------------------------------
// gelu(tanh)(g0) * g1 -> bf16
// ---------------------------------------------------------------------------
__global__ void gelumul_kernel(const float* __restrict__ g0,
                               const float* __restrict__ g1,
                               __bf16* __restrict__ o, int n) {
  const int i = blockIdx.x * blockDim.x + threadIdx.x;
  if (i >= n) return;
  const float a = g0[i];
  const float t = tanhf(0.7978845608028654f * (a + 0.044715f * a * a * a));
  o[i] = tobf(0.5f * a * (1.f + t) * g1[i]);
}

// ---------------------------------------------------------------------------
extern "C" void kernel_launch(void* const* d_in, const int* in_sizes, int n_in,
                              void* d_out, int out_size, void* d_ws,
                              size_t ws_size, hipStream_t stream) {
  (void)in_sizes; (void)n_in; (void)out_size; (void)ws_size;
  const int Sn = 2048;
  const int Mr = 2 * Sn;  // B*S = 4096

  const float* x      = (const float*)d_in[0];
  const int*   pad    = (const int*)d_in[1];
  const float* wqkv   = (const float*)d_in[2];
  const float* wo     = (const float*)d_in[3];
  const float* wi0    = (const float*)d_in[4];
  const float* wi1    = (const float*)d_in[5];
  const float* wom    = (const float*)d_in[6];
  const float* g_attn = (const float*)d_in[7];
  const float* g_mlp  = (const float*)d_in[8];

  char* w = (char*)d_ws;
  auto take = [&](size_t bytes) -> void* {
    void* p = (void*)w;
    w += (bytes + 255) & ~(size_t)255;
    return p;
  };

  // pre-transposed bf16 weights: [N, K]
  __bf16* wqkv_t = (__bf16*)take((size_t)2304 * 768 * 2);
  __bf16* wo_t   = (__bf16*)take((size_t)768 * 768 * 2);
  __bf16* wi0_t  = (__bf16*)take((size_t)1152 * 768 * 2);
  __bf16* wi1_t  = (__bf16*)take((size_t)1152 * 768 * 2);
  __bf16* wom_t  = (__bf16*)take((size_t)768 * 1152 * 2);
  __bf16* hn_b   = (__bf16*)take((size_t)Mr * 768 * 2);
  float*  qkvf   = (float*)take((size_t)Mr * 2304 * 4);
  __bf16* qb     = (__bf16*)take((size_t)Mr * 768 * 2);
  __bf16* kb     = (__bf16*)take((size_t)Mr * 768 * 2);
  __bf16* vb     = (__bf16*)take((size_t)Mr * 768 * 2);
  __bf16* attn_b = (__bf16*)take((size_t)Mr * 768 * 2);
  float*  x1f    = (float*)take((size_t)Mr * 768 * 4);
  __bf16* h2_b   = (__bf16*)take((size_t)Mr * 768 * 2);
  float*  g0f    = (float*)take((size_t)Mr * 1152 * 4);
  float*  g1f    = (float*)take((size_t)Mr * 1152 * 4);
  __bf16* hact_b = (__bf16*)take((size_t)Mr * 1152 * 2);

  // weight conversion + transpose: src [K,N] f32 -> dst [N,K] bf16
  const dim3 tb(32, 8);
  cvt_transpose_kernel<<<dim3(2304 / 32, 768 / 32),  tb, 0, stream>>>(wqkv, wqkv_t, 768, 2304);
  cvt_transpose_kernel<<<dim3(768 / 32,  768 / 32),  tb, 0, stream>>>(wo,   wo_t,   768, 768);
  cvt_transpose_kernel<<<dim3(1152 / 32, 768 / 32),  tb, 0, stream>>>(wi0,  wi0_t,  768, 1152);
  cvt_transpose_kernel<<<dim3(1152 / 32, 768 / 32),  tb, 0, stream>>>(wi1,  wi1_t,  768, 1152);
  cvt_transpose_kernel<<<dim3(768 / 32,  1152 / 32), tb, 0, stream>>>(wom,  wom_t,  1152, 768);

  // attention branch
  rmsnorm_kernel<<<Mr, 256, 0, stream>>>(x, g_attn, hn_b);
  gemm_bf16_kernel<<<dim3(2304 / 64, Mr / 64), 128, 0, stream>>>(
      hn_b, wqkv_t, nullptr, qkvf, Mr, 2304, 768);
  rope_kernel<<<(Mr * 12) / 256, 256, 0, stream>>>(qkvf, qb, kb, vb, Sn);
  attn_kernel<<<dim3(Sn / 64, 12, 2), 128, 0, stream>>>(qb, kb, vb, pad,
                                                        attn_b, Sn);
  gemm_bf16_kernel<<<dim3(768 / 64, Mr / 64), 128, 0, stream>>>(
      attn_b, wo_t, x, x1f, Mr, 768, 768);

  // MLP branch
  rmsnorm_kernel<<<Mr, 256, 0, stream>>>(x1f, g_mlp, h2_b);
  gemm_bf16_kernel<<<dim3(1152 / 64, Mr / 64), 128, 0, stream>>>(
      h2_b, wi0_t, nullptr, g0f, Mr, 1152, 768);
  gemm_bf16_kernel<<<dim3(1152 / 64, Mr / 64), 128, 0, stream>>>(
      h2_b, wi1_t, nullptr, g1f, Mr, 1152, 768);
  {
    const int n = Mr * 1152;
    gelumul_kernel<<<(n + 255) / 256, 256, 0, stream>>>(g0f, g1f, hact_b, n);
  }
  gemm_bf16_kernel<<<dim3(768 / 64, Mr / 64), 128, 0, stream>>>(
      hact_b, wom_t, x1f, (float*)d_out, Mr, 768, 1152);
}
